// AEM_7533372637631
// MI455X (gfx1250) — matrix operand order repaired and checked by reading, not compiled, not run on hardware
//
#include <hip/hip_runtime.h>
#include <hip/hip_fp16.h>
#include <stdint.h>

typedef __attribute__((ext_vector_type(16))) _Float16 v16h;
typedef __attribute__((ext_vector_type(8)))  _Float16 v8h;
typedef __attribute__((ext_vector_type(8)))  float    v8f;

#define B_DIM   4096
#define I_DIM   50
#define QW_DIM  20
#define E_DIM   128
#define H_DIM   8
#define AQ_OUT  (E_DIM * H_DIM)   // 1024

// ---------------------------------------------------------------------------
// Fast tanh: use gfx1250's hardware V_TANH_F32 (TRANS32 op) when the builtin
// exists; else fall back to libm. Removes the ~30-inst divergent libm expansion
// from the GEMM epilogues.
// ---------------------------------------------------------------------------
__device__ __forceinline__ float fast_tanh(float x) {
#if __has_builtin(__builtin_amdgcn_tanhf)
  return __builtin_amdgcn_tanhf(x);
#else
  return tanhf(x);
#endif
}

// ---------------------------------------------------------------------------
// Async global->LDS 16B copy (CDNA5 GLOBAL_LOAD_ASYNC_TO_LDS_B128, ASYNCcnt).
// Builtin signature (from hipcc diagnostic): first param is a pointer to a
// 16-byte int vector in the global address space; LDS destination is the
// AS(3) counterpart. Integer round-trip performs the addrspace conversion
// (generic LDS flat addr low 32 bits == LDS offset, ISA 10.2).
// ---------------------------------------------------------------------------
#if __has_builtin(__builtin_amdgcn_global_load_async_to_lds_b128)
#define HAVE_ASYNC_LDS 1
#else
#define HAVE_ASYNC_LDS 0
#endif

typedef int vi4 __attribute__((vector_size(4 * sizeof(int))));
typedef __attribute__((address_space(1))) vi4* as1_vi4_ptr;
typedef __attribute__((address_space(3))) vi4* as3_vi4_ptr;

__device__ __forceinline__ void copy16B_to_lds(const float* __restrict__ g,
                                               float* __restrict__ l) {
#if HAVE_ASYNC_LDS
  as1_vi4_ptr gp = (as1_vi4_ptr)(uintptr_t)g;
  as3_vi4_ptr lp = (as3_vi4_ptr)(uint32_t)(uintptr_t)l;
  __builtin_amdgcn_global_load_async_to_lds_b128(gp, lp, 0, 0);
#else
  *(float4*)l = *(const float4*)g;
#endif
}

__device__ __forceinline__ void wait_async_copies() {
#if HAVE_ASYNC_LDS
#if __has_builtin(__builtin_amdgcn_s_wait_asynccnt)
  __builtin_amdgcn_s_wait_asynccnt(0);
#else
  asm volatile("s_wait_asynccnt 0" ::: "memory");
#endif
#endif
}

// ---------------------------------------------------------------------------
// WMMA fragment loaders (f16, 16x16x32), per CDNA5 ISA 7.12.2 layouts.
// A (16x32, MxK): lane row = lane%16, kbase = 8*(lane/16);
//   halfwords [0..7]  -> K = kk + kbase + h
//   halfwords [8..15] -> K = kk + 16 + kbase + (h-8)
// B (32x16, KxN): lane col = lane%16, kbase = 8*(lane/16);
//   halfwords [0..7]  -> K = kk + kbase + h
//   halfwords [8..15] -> K = kk + 16 + kbase + (h-8)
// Both reduce to two 16-byte (b128) global loads per lane.
// ---------------------------------------------------------------------------
__device__ __forceinline__ v16h load_a_frag(const _Float16* __restrict__ A,
                                            int m0, int kk) {
  const int lane = threadIdx.x & 31;
  const int row  = lane & 15;
  const int kb   = (lane >> 4) << 3;           // 0 or 8
  const _Float16* p = A + (size_t)(m0 + row) * E_DIM + kk + kb;
  v8h lo = *(const v8h*)p;
  v8h hi = *(const v8h*)(p + 16);
  v16h a;
#pragma unroll
  for (int i = 0; i < 8; ++i) { a[i] = lo[i]; a[8 + i] = hi[i]; }
  return a;
}

__device__ __forceinline__ v16h load_b_frag(const _Float16* __restrict__ W,
                                            int n0, int kk) {
  // B[k,n] = W[n,k] for out = x @ W^T, W row-major (N x 128)
  const int lane = threadIdx.x & 31;
  const int n    = n0 + (lane & 15);
  const int ko   = (lane >> 4) << 3;           // 0 or 8
  const _Float16* p = W + (size_t)n * E_DIM + kk + ko;
  v8h lo = *(const v8h*)p;
  v8h hi = *(const v8h*)(p + 16);
  v16h b;
#pragma unroll
  for (int i = 0; i < 8; ++i) { b[i] = lo[i]; b[8 + i] = hi[i]; }
  return b;
}

// ---------------------------------------------------------------------------
// Kernel 0: fp32 -> f16 weight conversion (qp_w: 16384, aq_w: 131072 elems)
// ---------------------------------------------------------------------------
__global__ __launch_bounds__(256) void cvt_weights_kernel(
    const float* __restrict__ qp_w, const float* __restrict__ aq_w,
    _Float16* __restrict__ qp_w_h, _Float16* __restrict__ aq_w_h) {
  const int i = blockIdx.x * 256 + threadIdx.x;
  if (i < E_DIM * E_DIM)  qp_w_h[i] = (_Float16)qp_w[i];
  if (i < AQ_OUT * E_DIM) aq_w_h[i] = (_Float16)aq_w[i];
}

// ---------------------------------------------------------------------------
// Kernel 1: qmean[b,e] = mean_w word_emb[query_words[b,w], e]  (f16 output)
// Coalesced: 128 threads read consecutive floats of each gathered row.
// ---------------------------------------------------------------------------
__global__ __launch_bounds__(128) void qmean_kernel(
    const int* __restrict__ query_words, const float* __restrict__ word_emb,
    _Float16* __restrict__ qmean_h) {
  const int b = blockIdx.x;
  const int e = threadIdx.x;
  float s = 0.0f;
#pragma unroll
  for (int w = 0; w < QW_DIM; ++w) {
    const int idx = query_words[b * QW_DIM + w];
    s += word_emb[(size_t)idx * E_DIM + e];
  }
  qmean_h[(size_t)b * E_DIM + e] = (_Float16)(s * (1.0f / QW_DIM));
}

// ---------------------------------------------------------------------------
// Kernel 2: q = tanh(qmean @ qp_w^T + qp_b)   (B x 128) via WMMA f16->f32
// Block: 256 thr = 8 waves; wave w owns 16x16 N-tile w; 4 K-steps of 32.
// ---------------------------------------------------------------------------
__global__ __launch_bounds__(256) void qproj_kernel(
    const _Float16* __restrict__ qmean_h, const _Float16* __restrict__ qp_w_h,
    const float* __restrict__ qp_b,
    float* __restrict__ q_f32, _Float16* __restrict__ q_h) {
  const int wave = threadIdx.x >> 5;
  const int lane = threadIdx.x & 31;
  const int m0 = blockIdx.x * 16;
  const int n0 = wave * 16;
  v8f c = {};
#pragma unroll
  for (int kk = 0; kk < E_DIM; kk += 32) {
    v16h a = load_a_frag(qmean_h, m0, kk);
    v16h b = load_b_frag(qp_w_h, n0, kk);
    c = __builtin_amdgcn_wmma_f32_16x16x32_f16(false, a, false, b,
                                               (short)0, c, false, false);
  }
  const int n = n0 + (lane & 15);
  const float bias = qp_b[n];
#pragma unroll
  for (int r = 0; r < 8; ++r) {
    const int m = m0 + r + ((lane >> 4) << 3);
    const float v = fast_tanh(c[r] + bias);
    q_f32[(size_t)m * E_DIM + n] = v;
    q_h[(size_t)m * E_DIM + n]   = (_Float16)v;
  }
}

// ---------------------------------------------------------------------------
// Kernel 3: z = tanh(q @ aq_w^T + aq_b) (B x 1024) via WMMA, then fold red_w:
//   rvec[b,e] = sum_h z[b, e*8+h] * red_w[h]
// Grid (B/16, 8): block.y selects 128 of the 1024 output columns (=16 e's).
// ---------------------------------------------------------------------------
__global__ __launch_bounds__(256) void aproj_kernel(
    const _Float16* __restrict__ q_h, const _Float16* __restrict__ aq_w_h,
    const float* __restrict__ aq_b, const float* __restrict__ red_w,
    float* __restrict__ rvec) {
  __shared__ float zt[16][128];
  const int wave = threadIdx.x >> 5;
  const int lane = threadIdx.x & 31;
  const int m0 = blockIdx.x * 16;
  const int nt = blockIdx.y * 8 + wave;   // 16-wide column tile in [0,64)
  const int n0 = nt * 16;
  v8f c = {};
#pragma unroll
  for (int kk = 0; kk < E_DIM; kk += 32) {
    v16h a = load_a_frag(q_h, m0, kk);
    v16h b = load_b_frag(aq_w_h, n0, kk);
    c = __builtin_amdgcn_wmma_f32_16x16x32_f16(false, a, false, b,
                                               (short)0, c, false, false);
  }
  const int j  = n0 + (lane & 15);
  const int jl = (wave << 4) + (lane & 15);     // column within block's 128
  const float bias = aq_b[j];
#pragma unroll
  for (int r = 0; r < 8; ++r) {
    const int mr = r + ((lane >> 4) << 3);
    zt[mr][jl] = fast_tanh(c[r] + bias);
  }
  __syncthreads();
  // contract H=8 with red_w: thread (m, el) -> rvec[m0+m, blockIdx.y*16+el]
  const int m  = threadIdx.x >> 4;
  const int el = threadIdx.x & 15;
  float acc = 0.0f;
#pragma unroll
  for (int h = 0; h < H_DIM; ++h) acc += zt[m][el * H_DIM + h] * red_w[h];
  rvec[(size_t)(m0 + m) * E_DIM + blockIdx.y * 16 + el] = acc;
}

// ---------------------------------------------------------------------------
// Kernel 4: per-batch attention. One block per b.
//   scores[i] = <bought[i,:], rvec[b,:]>; softmax-with-mask per reference;
//   user = sum_i w[i] * bought[i,:]; out = 0.5*(q + user)
// bought gathered memory->LDS via async b128 copies (no VGPR round-trip);
// stride 132 floats keeps float4 alignment and de-conflicts banks.
// ---------------------------------------------------------------------------
#define ROW_STRIDE 132
__global__ __launch_bounds__(256) void attn_kernel(
    const int* __restrict__ user_bought_items,
    const float* __restrict__ item_emb,
    const float* __restrict__ masks,
    const float* __restrict__ rvec,
    const float* __restrict__ q_f32,
    float* __restrict__ out) {
  __shared__ float sb[I_DIM * ROW_STRIDE];
  __shared__ float srv[E_DIM];
  __shared__ float sred[I_DIM];
  __shared__ float swt[I_DIM];
  const int b = blockIdx.x;
  const int t = threadIdx.x;

  if (t < E_DIM) srv[t] = rvec[(size_t)b * E_DIM + t];

  // gather 50 rows x 128 floats; each row is 512B contiguous in item_emb
  for (int idx = t; idx < I_DIM * 32; idx += 256) {
    const int row  = idx >> 5;
    const int quad = idx & 31;
    const int item = user_bought_items[b * I_DIM + row];
    copy16B_to_lds(item_emb + (size_t)item * E_DIM + quad * 4,
                   &sb[row * ROW_STRIDE + quad * 4]);
  }
  wait_async_copies();
  __syncthreads();

  if (t < I_DIM) {
    float d = 0.0f;
    const float* rowp = &sb[t * ROW_STRIDE];
#pragma unroll 4
    for (int e = 0; e < E_DIM; ++e) d += rowp[e] * srv[e];
    sred[t] = d;
  }
  __syncthreads();

  float mx = -3.402823466e38f;
  for (int i = 0; i < I_DIM; ++i) mx = fmaxf(mx, sred[i]);  // broadcast reads
  if (t < I_DIM) swt[t] = expf(sred[t] - mx) * masks[b * I_DIM + t];
  __syncthreads();

  float den = 0.0f;
  for (int i = 0; i < I_DIM; ++i) den += swt[i];
  den = (den < 1e-7f) ? (den + 1.0f) : den;
  const float inv = 1.0f / den;

  if (t < E_DIM) {
    float u = 0.0f;
    for (int i = 0; i < I_DIM; ++i) u += swt[i] * sb[i * ROW_STRIDE + t];
    out[(size_t)b * E_DIM + t] =
        0.5f * (q_f32[(size_t)b * E_DIM + t] + u * inv);
  }
}

// ---------------------------------------------------------------------------
// Launch
// ---------------------------------------------------------------------------
extern "C" void kernel_launch(void* const* d_in, const int* in_sizes, int n_in,
                              void* d_out, int out_size, void* d_ws, size_t ws_size,
                              hipStream_t stream) {
  (void)in_sizes; (void)n_in; (void)out_size; (void)ws_size;
  const int*   user_bought_items = (const int*)d_in[0];
  /* d_in[1] = items : unused in the test-mode reference path            */
  const int*   query_words       = (const int*)d_in[2];
  const float* masks             = (const float*)d_in[3];
  const float* word_emb          = (const float*)d_in[4];
  const float* item_emb          = (const float*)d_in[5];
  const float* qp_w              = (const float*)d_in[6];
  const float* qp_b              = (const float*)d_in[7];
  const float* aq_w              = (const float*)d_in[8];
  const float* aq_b              = (const float*)d_in[9];
  const float* red_w             = (const float*)d_in[10];
  float* out = (float*)d_out;

  // workspace carve-up (all offsets 16B-aligned)
  char* ws = (char*)d_ws;
  _Float16* qp_w_h  = (_Float16*)ws; ws += (size_t)E_DIM * E_DIM * 2;     // 32 KB
  _Float16* aq_w_h  = (_Float16*)ws; ws += (size_t)AQ_OUT * E_DIM * 2;    // 256 KB
  _Float16* qmean_h = (_Float16*)ws; ws += (size_t)B_DIM * E_DIM * 2;     // 1 MB
  _Float16* q_h     = (_Float16*)ws; ws += (size_t)B_DIM * E_DIM * 2;     // 1 MB
  float*    q_f32   = (float*)ws;    ws += (size_t)B_DIM * E_DIM * 4;     // 2 MB
  float*    rvec    = (float*)ws;    ws += (size_t)B_DIM * E_DIM * 4;     // 2 MB

  cvt_weights_kernel<<<(AQ_OUT * E_DIM + 255) / 256, 256, 0, stream>>>(
      qp_w, aq_w, qp_w_h, aq_w_h);
  qmean_kernel<<<B_DIM, 128, 0, stream>>>(query_words, word_emb, qmean_h);
  qproj_kernel<<<B_DIM / 16, 256, 0, stream>>>(qmean_h, qp_w_h, qp_b, q_f32, q_h);
  aproj_kernel<<<dim3(B_DIM / 16, 8), 256, 0, stream>>>(q_h, aq_w_h, aq_b,
                                                        red_w, rvec);
  attn_kernel<<<B_DIM, 256, 0, stream>>>(user_bought_items, item_emb, masks,
                                         rvec, q_f32, out);
}
